// GraphWaveletTransform_46334107189751
// MI455X (gfx1250) — matrix operand order repaired and checked by reading, not compiled, not run on hardware
//
#include <hip/hip_runtime.h>

typedef __attribute__((ext_vector_type(8)))  float          v8f;
typedef __attribute__((ext_vector_type(2)))  float          v2f;
typedef __attribute__((ext_vector_type(16))) __bf16         v16bf;
typedef __attribute__((ext_vector_type(16))) unsigned short v16us;
typedef __attribute__((ext_vector_type(8)))  unsigned short v8us;

#define WMMA_F32(a, b, c) \
  __builtin_amdgcn_wmma_f32_16x16x4_f32(false, (a), false, (b), (short)0, (c), false, false)
#define WMMA_BF16(a, b, c) \
  __builtin_amdgcn_wmma_f32_16x16x32_bf16(false, (a), false, (b), (short)0, (c), false, false)

__device__ __forceinline__ unsigned short f32_to_bf16_rne(float x) {
  unsigned int u = __float_as_uint(x);
  u += 0x7FFFu + ((u >> 16) & 1u);
  return (unsigned short)(u >> 16);
}
__device__ __forceinline__ float bf16hi_to_f32(unsigned short h) {
  return __uint_as_float((unsigned int)h << 16);
}
// Reconstruct 4 fp32 values from packed bf16 hi/lo pairs at element offset off.
__device__ __forceinline__ float4 recon4(const unsigned short* hi, const unsigned short* lo,
                                         size_t off) {
  const uint2 h = *(const uint2*)(hi + off);
  const uint2 l = *(const uint2*)(lo + off);
  float4 r;
  r.x = __uint_as_float((h.x & 0xFFFFu) << 16) + __uint_as_float((l.x & 0xFFFFu) << 16);
  r.y = __uint_as_float(h.x & 0xFFFF0000u)     + __uint_as_float(l.x & 0xFFFF0000u);
  r.z = __uint_as_float((h.y & 0xFFFFu) << 16) + __uint_as_float((l.y & 0xFFFFu) << 16);
  r.w = __uint_as_float(h.y & 0xFFFF0000u)     + __uint_as_float(l.y & 0xFFFF0000u);
  return r;
}
// Build a v16bf WMMA fragment from two 16-byte LDS pieces.
__device__ __forceinline__ v16bf frag16(const unsigned short* p0, const unsigned short* p1) {
  const v8us a = *(const v8us*)p0;
  const v8us b = *(const v8us*)p1;
  v16us w;
#pragma unroll
  for (int i = 0; i < 8; ++i) { w[i] = a[i]; w[i + 8] = b[i]; }
  return __builtin_bit_cast(v16bf, w);
}

// ---------------------------------------------------------------------------
// Split fp32 -> (hi, lo) bf16 pair, elementwise (4 elements / thread).
// ---------------------------------------------------------------------------
__global__ void wavelet_split_bf16(const float* __restrict__ src,
                                   unsigned short* __restrict__ hi,
                                   unsigned short* __restrict__ lo, int n4) {
  const int i = blockIdx.x * blockDim.x + threadIdx.x;
  if (i >= n4) return;
  const float4 v = ((const float4*)src)[i];
  const float f[4] = {v.x, v.y, v.z, v.w};
  unsigned short h[4], l[4];
#pragma unroll
  for (int j = 0; j < 4; ++j) {
    h[j] = f32_to_bf16_rne(f[j]);
    l[j] = f32_to_bf16_rne(f[j] - bf16hi_to_f32(h[j]));
  }
  ((uint2*)hi)[i] = make_uint2((unsigned)h[0] | ((unsigned)h[1] << 16),
                               (unsigned)h[2] | ((unsigned)h[3] << 16));
  ((uint2*)lo)[i] = make_uint2((unsigned)l[0] | ((unsigned)l[1] << 16),
                               (unsigned)l[2] | ((unsigned)l[3] << 16));
}

// ---------------------------------------------------------------------------
// Batched bf16 transpose: D[n][k] = S[k][n] for hi and lo arrays (per slice).
// 64x64 tile through LDS. grid (16,16,32), 256 threads.
// ---------------------------------------------------------------------------
__global__ __launch_bounds__(256) void wavelet_transpose_bf16(
    const unsigned short* __restrict__ Shi, const unsigned short* __restrict__ Slo,
    unsigned short* __restrict__ Dhi, unsigned short* __restrict__ Dlo) {
  const size_t MS = 1024ull * 1024ull;
  const int bt = blockIdx.z;
  const unsigned short* shi = Shi + (size_t)bt * MS;
  const unsigned short* slo = Slo + (size_t)bt * MS;
  unsigned short* dhi = Dhi + (size_t)bt * MS;
  unsigned short* dlo = Dlo + (size_t)bt * MS;
  const int k0 = blockIdx.y * 64;  // source row block
  const int n0 = blockIdx.x * 64;  // source col block

  __shared__ unsigned short Lh[64][68];
  __shared__ unsigned short Ll[64][68];

  const int t = threadIdx.x;
  const int r = t >> 2, c = (t & 3) << 4;
  {
    const unsigned short* ph = shi + (size_t)(k0 + r) * 1024 + n0 + c;
    const unsigned short* pl = slo + (size_t)(k0 + r) * 1024 + n0 + c;
#pragma unroll
    for (int j = 0; j < 4; ++j) {
      *(uint2*)&Lh[r][c + 4 * j] = *(const uint2*)(ph + 4 * j);
      *(uint2*)&Ll[r][c + 4 * j] = *(const uint2*)(pl + 4 * j);
    }
  }
  __syncthreads();
  {
    unsigned int oh[8], ol[8];
#pragma unroll
    for (int j = 0; j < 8; ++j) {
      oh[j] = (unsigned)Lh[c + 2 * j][r] | ((unsigned)Lh[c + 2 * j + 1][r] << 16);
      ol[j] = (unsigned)Ll[c + 2 * j][r] | ((unsigned)Ll[c + 2 * j + 1][r] << 16);
    }
    unsigned short* qh = dhi + (size_t)(n0 + r) * 1024 + k0 + c;
    unsigned short* ql = dlo + (size_t)(n0 + r) * 1024 + k0 + c;
    *(uint4*)(qh)     = make_uint4(oh[0], oh[1], oh[2], oh[3]);
    *(uint4*)(qh + 8) = make_uint4(oh[4], oh[5], oh[6], oh[7]);
    *(uint4*)(ql)     = make_uint4(ol[0], ol[1], ol[2], ol[3]);
    *(uint4*)(ql + 8) = make_uint4(ol[4], ol[5], ol[6], ol[7]);
  }
}

// ---------------------------------------------------------------------------
// Split-bf16 squaring GEMM: C = A @ A (fp32 accuracy via hi/lo emulation).
// A-side from row-major hi/lo, B-side from pre-transposed hi/lo (so every
// fragment read is contiguous). Block: 256 threads (8 waves), tile 64x128,
// each wave a 32x32 quadrant (2x2 accumulators), K-chunk = 32, 3 bf16 WMMAs
// per tile pair per chunk. Epilogue writes next level's hi/lo directly.
// ---------------------------------------------------------------------------
__global__ __launch_bounds__(256) void wavelet_sq_gemm_bf16(
    const unsigned short* __restrict__ Ahi_all, const unsigned short* __restrict__ Alo_all,
    const unsigned short* __restrict__ Thi_all, const unsigned short* __restrict__ Tlo_all,
    unsigned short* __restrict__ Chi_all, unsigned short* __restrict__ Clo_all) {
  const size_t MS = 1024ull * 1024ull;
  const int bt = blockIdx.z;
  const unsigned short* Ahi = Ahi_all + (size_t)bt * MS;
  const unsigned short* Alo = Alo_all + (size_t)bt * MS;
  const unsigned short* Thi = Thi_all + (size_t)bt * MS;
  const unsigned short* Tlo = Tlo_all + (size_t)bt * MS;
  unsigned short* Chi = Chi_all + (size_t)bt * MS;
  unsigned short* Clo = Clo_all + (size_t)bt * MS;
  const int m0 = blockIdx.y * 64;
  const int n0 = blockIdx.x * 128;

  __shared__ unsigned short AsHi[64][32];    // 4 KB
  __shared__ unsigned short AsLo[64][32];    // 4 KB
  __shared__ unsigned short BsHi[128][32];   // 8 KB  ([n][k] of original)
  __shared__ unsigned short BsLo[128][32];   // 8 KB

  const int tid  = threadIdx.x;
  const int lane = tid & 31;
  const int wave = tid >> 5;
  const int wm = (wave >> 2) * 32;
  const int wn = (wave & 3) * 32;
  const int lh = lane >> 4;
  const int ll = lane & 15;

  v8f acc[2][2];
  for (int i = 0; i < 2; ++i)
    for (int j = 0; j < 2; ++j)
      for (int v = 0; v < 8; ++v) acc[i][j][v] = 0.0f;

  for (int k0 = 0; k0 < 1024; k0 += 32) {
    // Fill A tiles (64x32 bf16, row-major): 1 b128 per array per thread.
    {
      const int r = tid >> 2, c = (tid & 3) << 3;
      const size_t off = (size_t)(m0 + r) * 1024 + k0 + c;
      *(uint4*)&AsHi[r][c] = *(const uint4*)(Ahi + off);
      *(uint4*)&AsLo[r][c] = *(const uint4*)(Alo + off);
    }
    // Fill B tiles (128x32 bf16 from transposed matrix): 2 b128 per array.
    {
      const int r = tid >> 1, c = (tid & 1) << 4;
      const size_t off = (size_t)(n0 + r) * 1024 + k0 + c;
      *(uint4*)&BsHi[r][c]     = *(const uint4*)(Thi + off);
      *(uint4*)&BsHi[r][c + 8] = *(const uint4*)(Thi + off + 8);
      *(uint4*)&BsLo[r][c]     = *(const uint4*)(Tlo + off);
      *(uint4*)&BsLo[r][c + 8] = *(const uint4*)(Tlo + off + 8);
    }
    __syncthreads();

    // A frag (16-bit 16x32): lane half lh -> K groups {8*lh..8*lh+7, 16+8*lh..}
    v16bf ah[2], al[2], bh[2], bl[2];
#pragma unroll
    for (int tm = 0; tm < 2; ++tm) {
      const int m = wm + tm * 16 + ll;
      ah[tm] = frag16(&AsHi[m][8 * lh], &AsHi[m][16 + 8 * lh]);
      al[tm] = frag16(&AsLo[m][8 * lh], &AsLo[m][16 + 8 * lh]);
    }
    // B frag (16-bit 32x16): lane half lh -> K groups {8*lh..8*lh+7, 16+8*lh..}
#pragma unroll
    for (int tn = 0; tn < 2; ++tn) {
      const int n = wn + tn * 16 + ll;
      bh[tn] = frag16(&BsHi[n][8 * lh], &BsHi[n][16 + 8 * lh]);
      bl[tn] = frag16(&BsLo[n][8 * lh], &BsLo[n][16 + 8 * lh]);
    }
#pragma unroll
    for (int tm = 0; tm < 2; ++tm)
#pragma unroll
      for (int tn = 0; tn < 2; ++tn) {
        acc[tm][tn] = WMMA_BF16(ah[tm], bl[tn], acc[tm][tn]);  // hi*lo
        acc[tm][tn] = WMMA_BF16(al[tm], bh[tn], acc[tm][tn]);  // lo*hi
        acc[tm][tn] = WMMA_BF16(ah[tm], bh[tn], acc[tm][tn]);  // hi*hi
      }
    __syncthreads();
  }
  // Epilogue: split fp32 accumulators into next level's bf16 hi/lo.
#pragma unroll
  for (int tm = 0; tm < 2; ++tm)
#pragma unroll
    for (int tn = 0; tn < 2; ++tn)
#pragma unroll
      for (int v = 0; v < 8; ++v) {
        const float x = acc[tm][tn][v];
        const unsigned short h = f32_to_bf16_rne(x);
        const unsigned short l = f32_to_bf16_rne(x - bf16hi_to_f32(h));
        const int row = m0 + wm + tm * 16 + v + 8 * lh;
        const int col = n0 + wn + tn * 16 + ll;
        Chi[(size_t)row * 1024 + col] = h;
        Clo[(size_t)row * 1024 + col] = l;
      }
}

// ---------------------------------------------------------------------------
// Skinny GEMM (fp32 WMMA 16x16x4): D = maybe_abs((A - B) @ S), where A (and
// optional B) are reconstructed from bf16 hi/lo level arrays. S/D: [1024,64].
// ---------------------------------------------------------------------------
__global__ __launch_bounds__(256) void wavelet_skinny_gemm(
    const unsigned short* __restrict__ AhiAll, const unsigned short* __restrict__ AloAll,
    const unsigned short* __restrict__ BhiAll, const unsigned short* __restrict__ BloAll,
    const float* __restrict__ Sall, float* __restrict__ Dall, int doAbs) {
  const size_t MS = 1024ull * 1024ull;
  const int bt = blockIdx.y;
  const unsigned short* Ahi = AhiAll + (size_t)bt * MS;
  const unsigned short* Alo = AloAll + (size_t)bt * MS;
  const unsigned short* Bhi = BhiAll ? BhiAll + (size_t)bt * MS : (const unsigned short*)0;
  const unsigned short* Blo = BloAll ? BloAll + (size_t)bt * MS : (const unsigned short*)0;
  const float* S = Sall + (size_t)bt * (1024u * 64u);
  float* D = Dall + (size_t)bt * (1024u * 64u);
  const int m0 = blockIdx.x * 128;

  __shared__ float As[128][32];  // 16 KB
  __shared__ float Ss[32][64];   // 8 KB

  const int tid  = threadIdx.x;
  const int lane = tid & 31;
  const int wave = tid >> 5;
  const int lh = lane >> 4;
  const int ll = lane & 15;

  v8f acc[4];
  for (int t = 0; t < 4; ++t)
    for (int v = 0; v < 8; ++v) acc[t][v] = 0.0f;

  for (int k0 = 0; k0 < 1024; k0 += 32) {
    {
      const int r = tid >> 1, cb = (tid & 1) << 4;
      const size_t off = (size_t)(m0 + r) * 1024 + k0 + cb;
#pragma unroll
      for (int j = 0; j < 4; ++j) {
        float4 va = recon4(Ahi, Alo, off + 4 * j);
        if (Bhi) {
          const float4 vb = recon4(Bhi, Blo, off + 4 * j);
          va.x -= vb.x; va.y -= vb.y; va.z -= vb.z; va.w -= vb.w;
        }
        *(float4*)&As[r][cb + 4 * j] = va;
      }
    }
    {
      const int r = tid >> 3, cb = (tid & 7) << 3;
      const float* ps = S + (size_t)(k0 + r) * 64 + cb;
      *(float4*)&Ss[r][cb]     = *(const float4*)(ps);
      *(float4*)&Ss[r][cb + 4] = *(const float4*)(ps + 4);
    }
    __syncthreads();
#pragma unroll
    for (int kk = 0; kk < 32; kk += 4) {
      v2f a = *(const v2f*)&As[wave * 16 + ll][kk + 2 * lh];
#pragma unroll
      for (int t = 0; t < 4; ++t) {
        v2f b;
        b.x = Ss[kk + 2 * lh][t * 16 + ll];
        b.y = Ss[kk + 2 * lh + 1][t * 16 + ll];
        acc[t] = WMMA_F32(a, b, acc[t]);
      }
    }
    __syncthreads();
  }
#pragma unroll
  for (int t = 0; t < 4; ++t)
#pragma unroll
    for (int v = 0; v < 8; ++v) {
      float val = acc[t][v];
      if (doAbs) val = __builtin_fabsf(val);
      D[(size_t)(m0 + wave * 16 + v + 8 * lh) * 64 + t * 16 + ll] = val;
    }
}

// ---------------------------------------------------------------------------
// sc_i = |PX[i+1] - PX[i]|
// ---------------------------------------------------------------------------
__global__ void wavelet_absdiff(const float* __restrict__ a, const float* __restrict__ b,
                                float* __restrict__ o, int n) {
  const int i = blockIdx.x * blockDim.x + threadIdx.x;
  if (i < n) o[i] = __builtin_fabsf(a[i] - b[i]);
}

// ---------------------------------------------------------------------------
// Mean-pool [1024,64] over nodes, divide by sum(mask); one block per slice.
// ---------------------------------------------------------------------------
__global__ __launch_bounds__(256) void wavelet_pool(const float* __restrict__ feat,
                                                    const float* __restrict__ mask,
                                                    float* __restrict__ out, int kidx) {
  const int bt = blockIdx.x;
  const float* f = feat + (size_t)bt * (1024u * 64u);
  const float* m = mask + (size_t)bt * 1024u;
  __shared__ float part[4][64];
  __shared__ float mred[256];
  const int t = threadIdx.x;
  const int col = t & 63, rr = t >> 6;
  float s = 0.0f;
  for (int n = rr; n < 1024; n += 4) s += f[(size_t)n * 64 + col];
  part[rr][col] = s;
  float ms = 0.0f;
  for (int n = t; n < 1024; n += 256) ms += m[n];
  mred[t] = ms;
  __syncthreads();
  for (int st = 128; st > 0; st >>= 1) {
    if (t < st) mred[t] += mred[t + st];
    __syncthreads();
  }
  if (rr == 0) {
    const float tot = part[0][col] + part[1][col] + part[2][col] + part[3][col];
    out[(size_t)bt * 704 + kidx * 64 + col] = tot / mred[0];
  }
}

// ---------------------------------------------------------------------------
// Orchestration. Workspace (elements):
//   Lhi[0..3], Llo[0..3] : 8 x 32M bf16 (ushort)   = 512 MB   (power levels)
//   Thi, Tlo             : 2 x 32M ushort          = 128 MB   (transposed)
//   PX[0..4], SC[0..3], U: 10 x 32*1024*64 fp32    =  80 MB
// ---------------------------------------------------------------------------
extern "C" void kernel_launch(void* const* d_in, const int* in_sizes, int n_in,
                              void* d_out, int out_size, void* d_ws, size_t ws_size,
                              hipStream_t stream) {
  (void)in_sizes; (void)n_in; (void)out_size; (void)ws_size;
  const float* P    = (const float*)d_in[0];  // [8,4,1024,1024]
  const float* X    = (const float*)d_in[1];  // [8,4,1024,64]
  const float* mask = (const float*)d_in[2];  // [8,4,1024]
  float* out = (float*)d_out;                 // [8,4,704]

  unsigned short* us = (unsigned short*)d_ws;
  const size_t MATUS = 32ull * 1024ull * 1024ull;  // elements per level array
  unsigned short* Lhi[4];
  unsigned short* Llo[4];
  for (int i = 0; i < 4; ++i) {
    Lhi[i] = us + (size_t)i * MATUS;
    Llo[i] = us + (size_t)(4 + i) * MATUS;
  }
  unsigned short* Thi = us + 8 * MATUS;
  unsigned short* Tlo = us + 9 * MATUS;
  float* fws = (float*)(us + 10 * MATUS);
  const size_t SL = 32ull * 1024ull * 64ull;
  float* PX = fws;            // 5 slots
  float* SC = PX + 5 * SL;    // 4 slots
  float* U  = SC + 4 * SL;    // 1 slot

  const dim3 blk(256);

  // Split input P into bf16 hi/lo (level 0).
  {
    const int n4 = (int)(MATUS / 4);
    wavelet_split_bf16<<<dim3((n4 + 255) / 256), blk, 0, stream>>>(P, Lhi[0], Llo[0], n4);
  }
  // Squaring chain with split-bf16 WMMA: L1=P^2, L2=P^4, L3=P^8.
  for (int i = 0; i < 3; ++i) {
    wavelet_transpose_bf16<<<dim3(16, 16, 32), blk, 0, stream>>>(Lhi[i], Llo[i], Thi, Tlo);
    wavelet_sq_gemm_bf16<<<dim3(8, 16, 32), blk, 0, stream>>>(Lhi[i], Llo[i], Thi, Tlo,
                                                              Lhi[i + 1], Llo[i + 1]);
  }

  // PX chain: PX[i+1] = P^(2^i) @ PX[i]  (P^16 never materialized).
  const dim3 gsk(8, 32);
  typedef const unsigned short* cus;
  wavelet_skinny_gemm<<<gsk, blk, 0, stream>>>(Lhi[0], Llo[0], (cus)0, (cus)0, X,           PX,          0);
  wavelet_skinny_gemm<<<gsk, blk, 0, stream>>>(Lhi[0], Llo[0], (cus)0, (cus)0, PX,          PX + 1 * SL, 0);
  wavelet_skinny_gemm<<<gsk, blk, 0, stream>>>(Lhi[1], Llo[1], (cus)0, (cus)0, PX + 1 * SL, PX + 2 * SL, 0);
  wavelet_skinny_gemm<<<gsk, blk, 0, stream>>>(Lhi[2], Llo[2], (cus)0, (cus)0, PX + 2 * SL, PX + 3 * SL, 0);
  wavelet_skinny_gemm<<<gsk, blk, 0, stream>>>(Lhi[3], Llo[3], (cus)0, (cus)0, PX + 3 * SL, PX + 4 * SL, 0);

  // sc_i = |PX[i+1] - PX[i]|   (linearity: psi_i X = P^(2^(i+1))X - P^(2^i)X)
  const int n = (int)SL;
  for (int i = 0; i < 4; ++i)
    wavelet_absdiff<<<dim3((n + 255) / 256), blk, 0, stream>>>(PX + (size_t)(i + 1) * SL,
                                                               PX + (size_t)i * SL,
                                                               SC + (size_t)i * SL, n);

  // F0 = P^16 X ; F1_i = sc_i
  wavelet_pool<<<dim3(32), blk, 0, stream>>>(PX + 4 * SL, mask, out, 0);
  for (int i = 0; i < 4; ++i)
    wavelet_pool<<<dim3(32), blk, 0, stream>>>(SC + (size_t)i * SL, mask, out, 1 + i);

  // Second order: |psi_j @ sc_i| = |(L[j+1]-L[j]) @ sc_i|, reference order.
  const int pj[6] = {0, 0, 1, 0, 1, 2};
  const int pi[6] = {1, 2, 2, 3, 3, 3};
  for (int p = 0; p < 6; ++p) {
    wavelet_skinny_gemm<<<gsk, blk, 0, stream>>>(Lhi[pj[p] + 1], Llo[pj[p] + 1],
                                                 Lhi[pj[p]], Llo[pj[p]],
                                                 SC + (size_t)pi[p] * SL, U, 1);
    wavelet_pool<<<dim3(32), blk, 0, stream>>>(U, mask, out, 5 + p);
  }
}
